// Net_80985903334288
// MI455X (gfx1250) — matrix operand order repaired and checked by reading, not compiled, not run on hardware
//
#include <hip/hip_runtime.h>
#include <hip/hip_bf16.h>
#include <math.h>

typedef _Float16 h16;
typedef __attribute__((ext_vector_type(16))) _Float16 v16h;
typedef __attribute__((ext_vector_type(8)))  _Float16 v8h;
typedef __attribute__((ext_vector_type(8)))  float    v8f;

#define NPTS 131072L
#define BN_EPS 1e-5f

static inline unsigned cdivl(long a, int b) { return (unsigned)((a + b - 1) / b); }

// ---------------------------------------------------------------------------
// Weight convert: fp32 (Co x Ci) -> f16 zero-padded (Cop x Kp), row-major over K
// ---------------------------------------------------------------------------
__global__ void k_convert_w(const float* __restrict__ W, h16* __restrict__ W16,
                            int Co, int Ci, int Kp, int total) {
  int i = blockIdx.x * blockDim.x + threadIdx.x;
  if (i >= total) return;
  int r = i / Kp, c = i % Kp;
  float v = (r < Co && c < Ci) ? W[r * Ci + c] : 0.f;
  W16[i] = (h16)v;
}

__global__ void k_zero_f32(float* p, int n) {
  int i = blockIdx.x * blockDim.x + threadIdx.x;
  if (i < n) p[i] = 0.f;
}

// ---------------------------------------------------------------------------
// WMMA GEMM: OUT[N x Co] = act( concat(A0,A1,A2) @ W16^T + bias )
// Sources are f16 with padded leading dims (lda multiple of 16 halves) so each
// 16-half span is two aligned v8h loads. Concat boundaries (c0, c0+c1) are
// multiples of 32 and cStage is wave-uniform, so the source select is a scalar
// branch. W16 is f16 [Cop x Kp], zero-padded in K, so pad-column garbage
// multiplies zero weights. 256 threads = 8 wave32 waves; block tile 128x64;
// wave tile 32x32 (2x2 WMMA); LDS double-buffered A staging, 1 barrier/K-step.
// ---------------------------------------------------------------------------
__device__ __forceinline__ void stage_tile(
    h16* __restrict__ dst,            // &As[buf][0]
    int rStage, int cStage, long rowStage, int k0,
    const h16* A0, int c0, int lda0,
    const h16* A1, int c1, int lda1,
    const h16* A2, int c2, int lda2, int N)
{
  const int kT = k0 + cStage;         // wave-uniform
  v8h lo = {0,0,0,0,0,0,0,0};
  v8h hi2 = {0,0,0,0,0,0,0,0};
  if (rowStage < N) {
    const h16* p = nullptr;
    if (kT < c0)                p = A0 + rowStage * (long)lda0 + kT;
    else if (kT < c0 + c1)      p = A1 + rowStage * (long)lda1 + (kT - c0);
    else if (kT < c0 + c1 + c2) p = A2 + rowStage * (long)lda2 + (kT - c0 - c1);
    if (p) { lo = *(const v8h*)p; hi2 = *(const v8h*)(p + 8); }
  }
  h16* d = dst + rStage * 40 + cStage;
  *(v8h*)d       = lo;
  *(v8h*)(d + 8) = hi2;
}

__global__ __launch_bounds__(256)
void k_gemm_f16(const h16* __restrict__ A0, int c0, int lda0,
                const h16* __restrict__ A1, int c1, int lda1,
                const h16* __restrict__ A2, int c2, int lda2,
                const h16* __restrict__ W16, const float* __restrict__ bias,
                h16* __restrict__ OUT, int ldo,
                int N, int Co, int Kp, int doRelu)
{
  __shared__ alignas(32) h16 As[2][128 * 40];
  const int tid  = threadIdx.x;
  const int lane = tid & 31;
  const int wave = tid >> 5;
  const int l15  = lane & 15;
  const int hi   = lane >> 4;            // 0: lanes 0-15, 1: lanes 16-31
  const int mBase = (wave & 3) * 32;     // 4 waves along M
  const int nWave = (wave >> 2) * 32;    // 2 waves along N
  const long rowBlk = (long)blockIdx.x * 128;
  const int  nBlk   = blockIdx.y * 64;
  const int  totC   = c0 + c1 + c2;

  v8f acc[2][2];
  const v8f zero8 = {0.f,0.f,0.f,0.f,0.f,0.f,0.f,0.f};
  acc[0][0] = zero8; acc[0][1] = zero8; acc[1][0] = zero8; acc[1][1] = zero8;

  // wave-uniform column selection: waves 0-3 stage halves [0,16), waves 4-7 [16,32)
  const int  rStage   = tid & 127;
  const int  cStage   = (tid >> 7) * 16;
  const long rowStage = rowBlk + rStage;

  const int nT = Kp >> 5;                // number of 32-wide K tiles
  stage_tile(&As[0][0], rStage, cStage, rowStage, 0,
             A0,c0,lda0, A1,c1,lda1, A2,c2,lda2, N);

  for (int t = 0; t < nT; ++t) {
    __syncthreads();
    const int k0  = t << 5;
    const int buf = t & 1;

    // ---- stage next K tile into the alternate LDS buffer
    if (t + 1 < nT)
      stage_tile(&As[buf ^ 1][0], rStage, cStage, rowStage, k0 + 32,
                 A0,c0,lda0, A1,c1,lda1, A2,c2,lda2, N);
    // ---- prefetch two tiles ahead (global_prefetch_b8); select is wave-uniform
    {
      int kT = k0 + 64 + cStage;
      if (rowStage < N && kT < totC) {
        const h16* p;
        if (kT < c0)           p = A0 + rowStage * (long)lda0 + kT;
        else if (kT < c0 + c1) p = A1 + rowStage * (long)lda1 + (kT - c0);
        else                   p = A2 + rowStage * (long)lda2 + (kT - c0 - c1);
        __builtin_prefetch(p, 0, 0);
      }
    }

    union V16 { v16h v; v8h h[2]; };
    V16 af[2], bf[2];
    // A frag: ISA 16-bit A 16x32 layout. lanes 0-15: K {0..7,16..23} of row l15;
    // lanes 16-31: K {8..15,24..31}.
    #pragma unroll
    for (int i = 0; i < 2; ++i) {
      int mrow = mBase + 16 * i + l15;
      const h16* ap = &As[buf][mrow * 40 + hi * 8];
      af[i].h[0] = *(const v8h*)ap;
      af[i].h[1] = *(const v8h*)(ap + 16);
    }
    // B frag: lane<16 -> col n, K {0..7,16..23}; lane>=16 -> K {8..15,24..31}.
    #pragma unroll
    for (int j = 0; j < 2; ++j) {
      long n = nBlk + nWave + 16 * j + l15;
      const h16* bp = W16 + n * (long)Kp + k0 + hi * 8;
      bf[j].h[0] = *(const v8h*)bp;
      bf[j].h[1] = *(const v8h*)(bp + 16);
    }
    #pragma unroll
    for (int i = 0; i < 2; ++i)
      #pragma unroll
      for (int j = 0; j < 2; ++j)
        acc[i][j] = __builtin_amdgcn_wmma_f32_16x16x32_f16(
            false, af[i].v, false, bf[j].v, (short)0, acc[i][j], false, false);
  }

  // ---- epilogue: C layout VGPR v -> M = v + 8*hi, N = l15
  #pragma unroll
  for (int i = 0; i < 2; ++i) {
    #pragma unroll
    for (int j = 0; j < 2; ++j) {
      union { v8f v; float f[8]; } u; u.v = acc[i][j];
      int colg = nBlk + nWave + 16 * j + l15;
      if (colg < Co) {
        float b = bias[colg];
        #pragma unroll
        for (int v = 0; v < 8; ++v) {
          long rowg = rowBlk + mBase + 16 * i + v + 8 * hi;
          if (rowg < N) {
            float f = u.f[v] + b;
            if (doRelu) f = fmaxf(f, 0.f);
            OUT[rowg * (long)ldo + colg] = (h16)f;
          }
        }
      }
    }
  }
}

// ---------------------------------------------------------------------------
// sp1: h = center @ sp_w1^T + b   (3 -> 64)
// ---------------------------------------------------------------------------
__global__ void k_sp1(const float* __restrict__ x, const float* __restrict__ w,
                      const float* __restrict__ b, h16* __restrict__ out) {
  long r = (long)blockIdx.x * blockDim.x + threadIdx.x;
  if (r >= NPTS) return;
  float cx = x[r*18+0], cy = x[r*18+1], cz = x[r*18+2];
  #pragma unroll 8
  for (int o = 0; o < 64; ++o) {
    float v = b[o] + w[o*3+0]*cx + w[o*3+1]*cy + w[o*3+2]*cz;
    out[r*64 + o] = (h16)v;
  }
}

// ---------------------------------------------------------------------------
// BN stats: thread-per-column, 256 rows per block, register accumulation,
// two global atomic_add_f32 per thread. No integer division.
// ---------------------------------------------------------------------------
__global__ void k_bn_stats(const h16* __restrict__ h, float* __restrict__ acc,
                           int C, int ld) {
  int c = threadIdx.x;
  if (c >= C) return;
  long base = (long)blockIdx.x * 256;
  float s = 0.f, q = 0.f;
  for (int rr = 0; rr < 256; ++rr) {
    float v = (float)h[(base + rr) * (long)ld + c];
    s += v; q += v * v;
  }
  atomicAdd(&acc[c], s);
  atomicAdd(&acc[C + c], q);
}

__global__ void k_bn_final(const float* __restrict__ acc, const float* __restrict__ gamma,
                           const float* __restrict__ beta, float* __restrict__ scsh, int C) {
  int c = blockIdx.x * blockDim.x + threadIdx.x;
  if (c >= C) return;
  float mean = acc[c] * (1.f / (float)NPTS);
  float var  = acc[C + c] * (1.f / (float)NPTS) - mean * mean;
  float sc   = gamma[c] * rsqrtf(var + BN_EPS);
  scsh[c]     = sc;
  scsh[C + c] = beta[c] - mean * sc;
}

// thread-per-column, 8 rows per block; scale/shift loaded once per thread
__global__ void k_bn_apply_relu(h16* __restrict__ h, const float* __restrict__ scsh,
                                int C, int ld) {
  int c = threadIdx.x;
  if (c >= C) return;
  float sc = scsh[c], sh = scsh[C + c];
  long r0 = (long)blockIdx.x * 8;
  #pragma unroll
  for (int rr = 0; rr < 8; ++rr) {
    long a = (r0 + rr) * (long)ld + c;
    float v = (float)h[a] * sc + sh;
    h[a] = (h16)fmaxf(v, 0.f);
  }
}

// ---------------------------------------------------------------------------
// frc front: mean over 3 corner pairs of (pair @ frc_wc^T + bc)  -> N x 32
// ---------------------------------------------------------------------------
__global__ void k_frc0(const float* __restrict__ x, const float* __restrict__ w,
                       const float* __restrict__ b, h16* __restrict__ out) {
  long r = (long)blockIdx.x * blockDim.x + threadIdx.x;
  if (r >= NPTS) return;
  float a0=x[r*18+3],  a1=x[r*18+4],  a2=x[r*18+5];
  float b0=x[r*18+6],  b1=x[r*18+7],  b2=x[r*18+8];
  float d0=x[r*18+9],  d1=x[r*18+10], d2=x[r*18+11];
  #pragma unroll 4
  for (int o = 0; o < 32; ++o) {
    const float* wr = w + o*6;
    float s0 = wr[0]*a0+wr[1]*a1+wr[2]*a2 + wr[3]*b0+wr[4]*b1+wr[5]*b2; // (c1,c2)
    float s1 = wr[0]*b0+wr[1]*b1+wr[2]*b2 + wr[3]*d0+wr[4]*d1+wr[5]*d2; // (c2,c3)
    float s2 = wr[0]*d0+wr[1]*d1+wr[2]*d2 + wr[3]*a0+wr[4]*a1+wr[5]*a2; // (c3,c1)
    out[r*32 + o] = (h16)(b[o] + (s0 + s1 + s2) * (1.f/3.f));
  }
}

// ---------------------------------------------------------------------------
// kernel correlation + copy normal into s_in (cols 0..2 and 67..130, ld=144)
// ---------------------------------------------------------------------------
__global__ void k_kc_normal(const float* __restrict__ x, const float* __restrict__ kern,
                            h16* __restrict__ s_in) {
  long r = (long)blockIdx.x * blockDim.x + threadIdx.x;
  if (r >= NPTS) return;
  float nx=x[r*18+12], ny=x[r*18+13], nz=x[r*18+14];
  long base = r * 144;
  s_in[base+0]=(h16)nx; s_in[base+1]=(h16)ny; s_in[base+2]=(h16)nz;
  for (int o = 0; o < 64; ++o) {
    float acc = 0.f;
    #pragma unroll
    for (int k = 0; k < 4; ++k) {
      const float* kk = kern + (o*4 + k)*3;
      float dx=nx-kk[0], dy=ny-kk[1], dz=nz-kk[2];
      acc += expf(-(dx*dx + dy*dy + dz*dz));
    }
    s_in[base + 67 + o] = (h16)(acc * (1.f/32.f));   // 2*KC_K*4 = 32
  }
}

// ---------------------------------------------------------------------------
// agg_mean: thread-per-column, 8 rows per block; idx from x cols 15..17
// ---------------------------------------------------------------------------
__global__ void k_agg(const h16* __restrict__ f, const float* __restrict__ x,
                      h16* __restrict__ out, int C, int ld) {
  int c = threadIdx.x;
  if (c >= C) return;
  long r0 = (long)blockIdx.x * 8;
  #pragma unroll
  for (int rr = 0; rr < 8; ++rr) {
    long r = r0 + rr;
    long i0 = (long)(int)x[r*18+15];
    long i1 = (long)(int)x[r*18+16];
    long i2 = (long)(int)x[r*18+17];
    float v = (float)f[r*(long)ld + c] + (float)f[i0*(long)ld + c]
            + (float)f[i1*(long)ld + c] + (float)f[i2*(long)ld + c];
    out[r*(long)ld + c] = (h16)(0.25f * v);
  }
}

// ---------------------------------------------------------------------------
// Column-max over N (two stage): out6 is N x 1024 f16
// ---------------------------------------------------------------------------
__global__ void k_max_partial(const h16* __restrict__ a, float* __restrict__ pmax) {
  int col = blockIdx.x * blockDim.x + threadIdx.x;   // 0..1023
  long r0 = (long)blockIdx.y * 1024;
  float m = -3.4e38f;
  for (int rr = 0; rr < 1024; ++rr)
    m = fmaxf(m, (float)a[(r0 + rr) * 1024 + col]);
  pmax[(long)blockIdx.y * 1024 + col] = m;
}

__global__ void k_max_final(const float* __restrict__ pmax, float* __restrict__ mv) {
  int col = blockIdx.x * blockDim.x + threadIdx.x;
  float m = -3.4e38f;
  for (int c = 0; c < 128; ++c) m = fmaxf(m, pmax[c*1024 + col]);
  mv[col] = m;
}

// ---------------------------------------------------------------------------
// Head: 1024 -> 512 relu -> 256 relu -> 40 -> softmax (single block)
// ---------------------------------------------------------------------------
__global__ void k_head(const float* __restrict__ vin,
                       const float* __restrict__ w1, const float* __restrict__ b1,
                       const float* __restrict__ w2, const float* __restrict__ b2,
                       const float* __restrict__ w3, const float* __restrict__ b3,
                       float* __restrict__ out) {
  __shared__ float s_in[1024];
  __shared__ float s_h1[512];
  __shared__ float s_h2[256];
  __shared__ float s_lg[40];
  int t = threadIdx.x;
  for (int i = t; i < 1024; i += 256) s_in[i] = vin[i];
  __syncthreads();
  for (int o = t; o < 512; o += 256) {
    float a = b1[o]; const float* wr = w1 + (long)o * 1024;
    for (int k = 0; k < 1024; ++k) a += wr[k] * s_in[k];
    s_h1[o] = fmaxf(a, 0.f);
  }
  __syncthreads();
  for (int o = t; o < 256; o += 256) {
    float a = b2[o]; const float* wr = w2 + (long)o * 512;
    for (int k = 0; k < 512; ++k) a += wr[k] * s_h1[k];
    s_h2[o] = fmaxf(a, 0.f);
  }
  __syncthreads();
  if (t < 40) {
    float a = b3[t]; const float* wr = w3 + (long)t * 256;
    for (int k = 0; k < 256; ++k) a += wr[k] * s_h2[k];
    s_lg[t] = a;
  }
  __syncthreads();
  if (t == 0) {
    float m = -3.4e38f;
    for (int i = 0; i < 40; ++i) m = fmaxf(m, s_lg[i]);
    float ssum = 0.f;
    for (int i = 0; i < 40; ++i) { float e = expf(s_lg[i] - m); out[i] = e; ssum += e; }
    float inv = 1.f / ssum;
    for (int i = 0; i < 40; ++i) out[i] *= inv;
  }
}

// ---------------------------------------------------------------------------
extern "C" void kernel_launch(void* const* d_in, const int* in_sizes, int n_in,
                              void* d_out, int out_size, void* d_ws, size_t ws_size,
                              hipStream_t stream) {
  (void)in_sizes; (void)n_in; (void)out_size; (void)ws_size;
  const float* x      = (const float*)d_in[0];
  const float* sp_w1  = (const float*)d_in[1];
  const float* sp_b1  = (const float*)d_in[2];
  const float* sp_g1  = (const float*)d_in[3];
  const float* sp_be1 = (const float*)d_in[4];
  const float* sp_w2  = (const float*)d_in[5];
  const float* sp_b2  = (const float*)d_in[6];
  const float* frc_wc = (const float*)d_in[7];
  const float* frc_bc = (const float*)d_in[8];
  const float* frc_w3 = (const float*)d_in[9];
  const float* frc_b3 = (const float*)d_in[10];
  const float* frc_w4 = (const float*)d_in[11];
  const float* frc_b4 = (const float*)d_in[12];
  const float* kc_k   = (const float*)d_in[13];
  const float* st_w1  = (const float*)d_in[14];
  const float* st_b1  = (const float*)d_in[15];
  const float* st_g1  = (const float*)d_in[16];
  const float* st_be1 = (const float*)d_in[17];
  const float* st_w2  = (const float*)d_in[18];
  const float* st_b2  = (const float*)d_in[19];
  const float* c1_w   = (const float*)d_in[20];
  const float* c1_b   = (const float*)d_in[21];
  const float* a1_w   = (const float*)d_in[22];
  const float* a1_b   = (const float*)d_in[23];
  const float* c2_w   = (const float*)d_in[24];
  const float* c2_b   = (const float*)d_in[25];
  const float* a2_w   = (const float*)d_in[26];
  const float* a2_b   = (const float*)d_in[27];
  const float* net_w  = (const float*)d_in[28];
  const float* net_b  = (const float*)d_in[29];
  const float* m2_w   = (const float*)d_in[30];
  const float* m2_b   = (const float*)d_in[31];
  const float* m3_w1  = (const float*)d_in[32];
  const float* m3_b1  = (const float*)d_in[33];
  const float* m3_w2  = (const float*)d_in[34];
  const float* m3_b2  = (const float*)d_in[35];
  const float* m3_w3  = (const float*)d_in[36];
  const float* m3_b3  = (const float*)d_in[37];

  const long N = NPTS;
  char* base = (char*)d_ws; size_t off = 0;
  auto alloc = [&](size_t bytes) -> void* {
    void* p = base + off; off = (off + bytes + 255) & ~(size_t)255; return p;
  };

  // f16 padded weights (Cop x Kp)
  h16* w16_sp2 = (h16*)alloc((size_t)64*64*2);
  h16* w16_f3  = (h16*)alloc((size_t)64*32*2);
  h16* w16_f4  = (h16*)alloc((size_t)64*64*2);
  h16* w16_st1 = (h16*)alloc((size_t)192*160*2);
  h16* w16_st2 = (h16*)alloc((size_t)192*160*2);
  h16* w16_c1  = (h16*)alloc((size_t)256*224*2);
  h16* w16_a1  = (h16*)alloc((size_t)256*160*2);
  h16* w16_c2  = (h16*)alloc((size_t)512*512*2);
  h16* w16_a2  = (h16*)alloc((size_t)512*256*2);
  h16* w16_net = (h16*)alloc((size_t)1024*1024*2);
  h16* w16_m2  = (h16*)alloc((size_t)1024*1792*2);
  // f16 activations (leading dims padded to multiples of 16 halves)
  h16* h_sp = (h16*)alloc((size_t)N*64*2);
  h16* y    = (h16*)alloc((size_t)N*64*2);
  h16* frc0 = (h16*)alloc((size_t)N*32*2);
  h16* frc1 = (h16*)alloc((size_t)N*64*2);
  h16* s_in = (h16*)alloc((size_t)N*144*2);   // 131 cols, ld 144
  h16* h_st = (h16*)alloc((size_t)N*144*2);
  h16* s    = (h16*)alloc((size_t)N*144*2);
  h16* aggs = (h16*)alloc((size_t)N*144*2);
  h16* out1 = (h16*)alloc((size_t)N*256*2);
  h16* out2 = (h16*)alloc((size_t)N*256*2);
  h16* agg2 = (h16*)alloc((size_t)N*256*2);
  h16* out3 = (h16*)alloc((size_t)N*512*2);
  h16* out4 = (h16*)alloc((size_t)N*512*2);
  h16* out5 = (h16*)alloc((size_t)N*1024*2);
  h16* out6 = (h16*)alloc((size_t)N*1024*2);
  // fp32 small scratch
  float* acc_sp  = (float*)alloc(128*4);
  float* scsh_sp = (float*)alloc(128*4);
  float* acc_st  = (float*)alloc(262*4);
  float* scsh_st = (float*)alloc(262*4);
  float* pmax    = (float*)alloc((size_t)128*1024*4);
  float* maxvec  = (float*)alloc(1024*4);

  auto conv = [&](const float* W, h16* W16, int Co, int Ci, int Cop, int Kp) {
    int tot = Cop * Kp;
    k_convert_w<<<cdivl(tot,256),256,0,stream>>>(W, W16, Co, Ci, Kp, tot);
  };
  conv(sp_w2,  w16_sp2,  64,  64,  64,  64);
  conv(frc_w3, w16_f3,   64,  32,  64,  32);
  conv(frc_w4, w16_f4,   64,  64,  64,  64);
  conv(st_w1,  w16_st1, 131, 131, 192, 160);
  conv(st_w2,  w16_st2, 131, 131, 192, 160);
  conv(c1_w,   w16_c1,  256, 195, 256, 224);
  conv(a1_w,   w16_a1,  256, 131, 256, 160);
  conv(c2_w,   w16_c2,  512, 512, 512, 512);
  conv(a2_w,   w16_a2,  512, 256, 512, 256);
  conv(net_w,  w16_net,1024,1024,1024,1024);
  conv(m2_w,   w16_m2, 1024,1792,1024,1792);

  auto gemm = [&](const h16* A0,int c0_,int lda0,const h16* A1,int c1_,int lda1,
                  const h16* A2,int c2_,int lda2,
                  const h16* W16,const float* bias_, h16* OUT,int ldo,int Co,int Kp,int relu){
    dim3 g(cdivl(N,128), (unsigned)((Co + 63) / 64));
    k_gemm_f16<<<g,256,0,stream>>>(A0,c0_,lda0,A1,c1_,lda1,A2,c2_,lda2,
                                   W16,bias_,OUT,ldo,(int)N,Co,Kp,relu);
  };
  auto blkC = [](int C){ return (unsigned)(((C + 63) / 64) * 64); };

  // ---- sp path
  k_zero_f32<<<1,256,0,stream>>>(acc_sp,128);
  k_zero_f32<<<2,256,0,stream>>>(acc_st,262);
  k_sp1<<<cdivl(N,256),256,0,stream>>>(x, sp_w1, sp_b1, h_sp);
  k_bn_stats<<<512,blkC(64),0,stream>>>(h_sp, acc_sp, 64, 64);
  k_bn_final<<<1,64,0,stream>>>(acc_sp, sp_g1, sp_be1, scsh_sp, 64);
  k_bn_apply_relu<<<cdivl(N,8),blkC(64),0,stream>>>(h_sp, scsh_sp, 64, 64);
  gemm(h_sp,64,64, nullptr,0,0, nullptr,0,0, w16_sp2, sp_b2, y,64, 64,64, 0);
  // ---- frc path
  k_frc0<<<cdivl(N,256),256,0,stream>>>(x, frc_wc, frc_bc, frc0);
  gemm(frc0,32,32, nullptr,0,0, nullptr,0,0, w16_f3, frc_b3, frc1,64, 64,32, 1);
  gemm(frc1,64,64, nullptr,0,0, nullptr,0,0, w16_f4, frc_b4, s_in+3,144, 64,64, 0);
  // ---- kc + normal into s_in
  k_kc_normal<<<cdivl(N,256),256,0,stream>>>(x, kc_k, s_in);
  // ---- st path
  gemm(s_in,131,144, nullptr,0,0, nullptr,0,0, w16_st1, st_b1, h_st,144, 131,160, 0);
  k_bn_stats<<<512,blkC(131),0,stream>>>(h_st, acc_st, 131, 144);
  k_bn_final<<<1,256,0,stream>>>(acc_st, st_g1, st_be1, scsh_st, 131);
  k_bn_apply_relu<<<cdivl(N,8),blkC(131),0,stream>>>(h_st, scsh_st, 131, 144);
  gemm(h_st,131,144, nullptr,0,0, nullptr,0,0, w16_st2, st_b2, s,144, 131,160, 0);
  // ---- aggregation + big GEMM chain (concats read in place)
  k_agg<<<cdivl(N,8),blkC(131),0,stream>>>(s, x, aggs, 131, 144);
  gemm(y,64,64, s,131,144, nullptr,0,0, w16_c1, c1_b, out1,256, 256,224, 0);
  gemm(aggs,131,144, nullptr,0,0, nullptr,0,0, w16_a1, a1_b, out2,256, 256,160, 0);
  gemm(out1,256,256, out2,256,256, nullptr,0,0, w16_c2, c2_b, out3,512, 512,512, 0);
  k_agg<<<cdivl(N,8),blkC(256),0,stream>>>(out2, x, agg2, 256, 256);
  gemm(agg2,256,256, nullptr,0,0, nullptr,0,0, w16_a2, a2_b, out4,512, 512,256, 0);
  gemm(out3,512,512, out4,512,512, nullptr,0,0, w16_net, net_b, out5,1024, 1024,1024, 0);
  gemm(out5,1024,1024, out3,512,512, out1,256,256, w16_m2, m2_b, out6,1024, 1024,1792, 0);
  // ---- global max + head
  dim3 gm(4,128);
  k_max_partial<<<gm,256,0,stream>>>(out6, pmax);
  k_max_final<<<4,256,0,stream>>>(pmax, maxvec);
  k_head<<<1,256,0,stream>>>(maxvec, m3_w1,m3_b1, m3_w2,m3_b2, m3_w3,m3_b3, (float*)d_out);
}
